// ESN_83502754169067
// MI455X (gfx1250) — matrix operand order repaired and checked by reading, not compiled, not run on hardware
//
#include <hip/hip_runtime.h>

#define D_ 512
#define R_ 2048
#define T_ 128
#define L_ 64
#define G4_ 2048
#define NWIN_ 64
#define NWG_ 64
#define TPB_ 256
#define NTH_ (NWG_ * TPB_)
#define STEPS_ (T_ + L_ - 1) /* 191 */
#define RS_ 520              /* padded LDS row stride in bf16 (512 + 8) */
#define SMEM_BYTES_ ((16 + 128) * RS_ * 2) /* A slice (16 rows) + B slice (128 rows) */

typedef __attribute__((ext_vector_type(16))) __bf16 bf16x16;
typedef __attribute__((ext_vector_type(8)))  __bf16 bf16x8;
typedef __attribute__((ext_vector_type(8)))  float  f32x8;

// ---- workspace byte offsets ----
#define WS_BAR 0
#define WS_BUF 256                               // float[(T+L)*D]
#define WS_H   (WS_BUF + (T_ + L_) * D_ * 4)     // float[64*512]
#define WS_C   (WS_H + NWIN_ * D_ * 4)           // float[64*512]
#define WS_G   (WS_C + NWIN_ * D_ * 4)           // float[64*2048]
#define WS_PX  (WS_G + NWIN_ * G4_ * 4)          // float[(T+L)*2048]
#define WS_RB  (WS_PX + (T_ + L_) * G4_ * 4)     // float[2048]
#define WS_HBF (WS_RB + R_ * 4)                  // bf16[64*512]
#define WS_WHH (WS_HBF + NWIN_ * D_ * 2)         // bf16[2048*512]

__device__ __forceinline__ unsigned short f2bf(float f) {
  union { float f; unsigned u; } v; v.f = f;
  unsigned r = v.u + 0x7FFFu + ((v.u >> 16) & 1u); // RNE
  return (unsigned short)(r >> 16);
}
__device__ __forceinline__ float sigm(float x) { return 1.0f / (1.0f + __expf(-x)); }
__device__ __forceinline__ float tanh_f(float x) {
  float e = __expf(2.0f * x);
  return 1.0f - 2.0f / (e + 1.0f); // inf-safe: ->1 / ->-1
}

// sense-reversing device-wide barrier (self-resetting)
__device__ __forceinline__ void gbar(unsigned* cnt, unsigned* gen) {
  __syncthreads();
  if (threadIdx.x == 0) {
    __threadfence();
    unsigned g = __hip_atomic_load(gen, __ATOMIC_RELAXED, __HIP_MEMORY_SCOPE_AGENT);
    unsigned a = __hip_atomic_fetch_add(cnt, 1u, __ATOMIC_ACQ_REL, __HIP_MEMORY_SCOPE_AGENT);
    if (a == (unsigned)(NWG_ - 1)) {
      __hip_atomic_store(cnt, 0u, __ATOMIC_RELAXED, __HIP_MEMORY_SCOPE_AGENT);
      __hip_atomic_store(gen, g + 1u, __ATOMIC_RELEASE, __HIP_MEMORY_SCOPE_AGENT);
    } else {
      while (__hip_atomic_load(gen, __ATOMIC_ACQUIRE, __HIP_MEMORY_SCOPE_AGENT) == g) {
        __builtin_amdgcn_s_sleep(1);
      }
    }
    __threadfence();
  }
  __syncthreads();
}

__global__ void esn_bar_init(unsigned* bar) { bar[0] = 0u; bar[1] = 0u; }

__global__ __launch_bounds__(TPB_, 1) void esn_main(
    const float* __restrict__ x,   const float* __restrict__ Wih,
    const float* __restrict__ Whh, const float* __restrict__ bih,
    const float* __restrict__ bhh, const float* __restrict__ W1,
    const float* __restrict__ b1,  const float* __restrict__ W2,
    const float* __restrict__ b2,  float* __restrict__ out,
    unsigned char* __restrict__ ws) {
  extern __shared__ unsigned char smem[];
  unsigned short* Alds = (unsigned short*)smem;            // [16][RS_]
  unsigned short* Blds = Alds + 16 * RS_;                  // [128][RS_]

  unsigned* bar = (unsigned*)(ws + WS_BAR);
  float* buf = (float*)(ws + WS_BUF);
  float* H   = (float*)(ws + WS_H);
  float* C   = (float*)(ws + WS_C);
  float* G   = (float*)(ws + WS_G);
  float* PX  = (float*)(ws + WS_PX);
  float* RB  = (float*)(ws + WS_RB);
  unsigned short* Hbf   = (unsigned short*)(ws + WS_HBF);
  unsigned short* Whhbf = (unsigned short*)(ws + WS_WHH);

  const int gtid = blockIdx.x * TPB_ + threadIdx.x;
  const int lane = threadIdx.x & 31;
  const int wave = gtid >> 5;              // 0..511 (global, used by phase C2)
  const int wv   = threadIdx.x >> 5;       // 0..7   (wave in WG)

  // tile ownership: WG -> one M-tile x 8 N-tiles
  const int mt = blockIdx.x & 3;                 // 0..3
  const int ntbase = (blockIdx.x >> 2) * 8;      // 0,8,...,120
  const int nt = ntbase + wv;                    // 0..127
  const int lh = lane & 15, hi = lane >> 4;

  // ---------------- init ----------------
  for (int i = gtid; i < (T_ + L_) * D_; i += NTH_)
    buf[i] = (i < T_ * D_) ? x[i] : 0.0f;
  for (int i = gtid; i < NWIN_ * D_; i += NTH_) { H[i] = 0.f; C[i] = 0.f; Hbf[i] = 0; }
  for (int i = gtid; i < G4_ * D_; i += NTH_) Whhbf[i] = f2bf(Whh[i]);
  { // PX[s][n] = (Wih @ x_s)[n] for s = 0..127   (8 lanes per output, weights in regs)
    const int n = gtid >> 3, part = gtid & 7;
    const float* wr = Wih + n * D_ + part * 64;
    float w[64];
#pragma unroll
    for (int u = 0; u < 64; ++u) w[u] = wr[u];
    for (int sp = 0; sp < T_; ++sp) {
      const float* xv = x + sp * D_ + part * 64;
      float sm = 0.f;
#pragma unroll
      for (int u = 0; u < 64; ++u) sm = fmaf(w[u], xv[u], sm);
      sm += __shfl_xor(sm, 1, 32); sm += __shfl_xor(sm, 2, 32); sm += __shfl_xor(sm, 4, 32);
      if (part == 0) PX[sp * G4_ + n] = sm;
    }
  }
  gbar(bar, bar + 1);

  // ---- one-time: stage this WG's B slice (Whhbf rows [ntbase*16, ntbase*16+128)) in LDS ----
  for (int i = threadIdx.x; i < 128 * 64; i += TPB_) {  // 64 uint4 per row
    const int row = i >> 6, c16 = i & 63;
    *(uint4*)(Blds + row * RS_ + c16 * 8) =
        *(const uint4*)(Whhbf + (ntbase * 16 + row) * D_ + c16 * 8);
  }

  const unsigned short* arow = Alds + lh * RS_;            // A row for this lane
  const unsigned short* brow = Blds + (wv * 16 + lh) * RS_; // B row for this lane

  for (int s = 0; s < STEPS_; ++s) {
    // ---- Phase A: stage A slice (Hbf rows [mt*16, mt*16+16)) then WMMA from LDS ----
    for (int i = threadIdx.x; i < 16 * 64; i += TPB_) {   // 4 uint4 per thread
      const int row = i >> 6, c16 = i & 63;
      *(uint4*)(Alds + row * RS_ + c16 * 8) =
          *(const uint4*)(Hbf + (mt * 16 + row) * D_ + c16 * 8);
    }
    __syncthreads();

    // G(64x2048) = Hbf @ WhhbfT : 16 x v_wmma_f32_16x16x32_bf16 per wave, operands from LDS
    f32x8 acc = {0.f, 0.f, 0.f, 0.f, 0.f, 0.f, 0.f, 0.f};
#pragma unroll
    for (int kk = 0; kk < 16; ++kk) {
      const int ka = kk * 32 + hi * 8;   // A: lanes0-15 K{0..7,16..23}, lanes16-31 K{8..15,24..31}
      const int kb = kk * 32 + hi * 8;   // B: lanes0-15 K{0..7,16..23}, lanes16-31 K{8..15,24..31}
      bf16x8 a0 = *(const bf16x8*)(arow + ka);
      bf16x8 a1 = *(const bf16x8*)(arow + ka + 16);
      bf16x8 c0 = *(const bf16x8*)(brow + kb);
      bf16x8 c1 = *(const bf16x8*)(brow + kb + 16);
      bf16x16 av, bv;
#pragma unroll
      for (int e = 0; e < 8; ++e) { av[e] = a0[e]; av[e + 8] = a1[e]; bv[e] = c0[e]; bv[e + 8] = c1[e]; }
      acc = __builtin_amdgcn_wmma_f32_16x16x32_bf16(false, av, false, bv,
                                                    (short)0, acc, false, false);
    }
    { // C/D layout: VGPR r -> M = r + 8*hi, N = lane&15
      float* gout = G + (mt * 16 + hi * 8) * G4_ + nt * 16 + lh;
#pragma unroll
      for (int r = 0; r < 8; ++r) gout[r * G4_] = acc[r];
    }
    // Phase A tail: PX[s] = Wih @ buf[s] for autoregressive inputs (buf[s] from prev C2)
    if (s >= T_) {
      const int n = gtid >> 3, part = gtid & 7;
      const float* xv = buf + s * D_ + part * 64;
      const float* wr = Wih + n * D_ + part * 64;
      float sm = 0.f;
#pragma unroll 16
      for (int u = 0; u < 64; ++u) sm = fmaf(wr[u], xv[u], sm);
      sm += __shfl_xor(sm, 1, 32); sm += __shfl_xor(sm, 2, 32); sm += __shfl_xor(sm, 4, 32);
      if (part == 0) PX[s * G4_ + n] = sm;
    }
    gbar(bar, bar + 1);

    // ---- Phase B: LSTM cell update for active windows j in [max(0,s-127), min(63,s)] ----
    {
      const int lo = (s >= T_) ? (s - (T_ - 1)) : 0;
      const float* px = PX + s * G4_;
      for (int e = gtid; e < NWIN_ * D_; e += NTH_) {
        const int j = e >> 9, u = e & (D_ - 1);
        if (j >= lo && j <= s) {
          const float* gr = G + j * G4_;
          float gi = gr[u]          + px[u]          + bih[u]          + bhh[u];
          float gf = gr[u + 512]    + px[u + 512]    + bih[u + 512]    + bhh[u + 512];
          float gg = gr[u + 1024]   + px[u + 1024]   + bih[u + 1024]   + bhh[u + 1024];
          float go = gr[u + 1536]   + px[u + 1536]   + bih[u + 1536]   + bhh[u + 1536];
          float c = sigm(gf) * C[e] + sigm(gi) * tanh_f(gg);
          float h = sigm(go) * tanh_f(c);
          C[e] = c; H[e] = h; Hbf[e] = f2bf(h);
        }
      }
    }

    if (s >= T_ - 1) { // window jd = s-127 just finished -> MLP -> buf[s+1] (= output row jd)
      gbar(bar, bar + 1);
      const int jd = s - (T_ - 1);
      { // C1: RB = leaky_relu(W1 @ h_jd + b1), 8 lanes per output
        const int n = gtid >> 3, part = gtid & 7;
        const float* hv = H + jd * D_ + part * 64;
        const float* wr = W1 + n * D_ + part * 64;
        float sm = 0.f;
#pragma unroll 16
        for (int u = 0; u < 64; ++u) sm = fmaf(wr[u], hv[u], sm);
        sm += __shfl_xor(sm, 1, 32); sm += __shfl_xor(sm, 2, 32); sm += __shfl_xor(sm, 4, 32);
        if (part == 0) {
          float v = sm + b1[n];
          RB[n] = (v > 0.f) ? v : 0.01f * v;
        }
      }
      gbar(bar, bar + 1);
      { // C2: y = W2 @ RB + b2 ; one wave per output element d
        const int d = wave; // 0..511
        const float* wr = W2 + d * R_;
        float sm = 0.f;
#pragma unroll 8
        for (int t = 0; t < 64; ++t) sm = fmaf(wr[lane + 32 * t], RB[lane + 32 * t], sm);
        sm += __shfl_xor(sm, 1, 32);  sm += __shfl_xor(sm, 2, 32);
        sm += __shfl_xor(sm, 4, 32);  sm += __shfl_xor(sm, 8, 32);
        sm += __shfl_xor(sm, 16, 32);
        if (lane == 0) {
          float y = sm + b2[d];
          buf[(s + 1) * D_ + d] = y;
          out[jd * D_ + d] = y; // output [1,L,D] is exactly buf[128..191] flat
        }
      }
    }
    gbar(bar, bar + 1);
  }
}

extern "C" void kernel_launch(void* const* d_in, const int* in_sizes, int n_in,
                              void* d_out, int out_size, void* d_ws, size_t ws_size,
                              hipStream_t stream) {
  (void)in_sizes; (void)n_in; (void)out_size; (void)ws_size;
  const float* x   = (const float*)d_in[0];
  const float* Wih = (const float*)d_in[1];
  const float* Whh = (const float*)d_in[2];
  const float* bih = (const float*)d_in[3];
  const float* bhh = (const float*)d_in[4];
  const float* W1  = (const float*)d_in[5];
  const float* b1  = (const float*)d_in[6];
  const float* W2  = (const float*)d_in[7];
  const float* b2  = (const float*)d_in[8];
  float* out = (float*)d_out;
  unsigned char* ws = (unsigned char*)d_ws;

  esn_bar_init<<<dim3(1), dim3(1), 0, stream>>>((unsigned*)ws);
  esn_main<<<dim3(NWG_), dim3(TPB_), SMEM_BYTES_, stream>>>(x, Wih, Whh, bih, bhh,
                                                            W1, b1, W2, b2, out, ws);
}
